// DGC_backbone_12472585028248
// MI455X (gfx1250) — matrix operand order repaired and checked by reading, not compiled, not run on hardware
//
#include <hip/hip_runtime.h>
#include <hip/hip_bf16.h>

typedef __attribute__((ext_vector_type(16))) _Float16 v16h;
typedef __attribute__((ext_vector_type(8)))  float    v8f;

// ---------------------------------------------------------------------------
// Fast, branch-free transcendentals (serial GRU chain is latency-bound; the
// precise libm paths emit v_div_scale/v_div_fmas sequences and divergent
// slow-path branches which we must not pay per time-step).
// ---------------------------------------------------------------------------
__device__ __forceinline__ float fast_rcp(float x) {
  return __builtin_amdgcn_rcpf(x);
}
__device__ __forceinline__ float fast_sigmoid(float x) {
  return fast_rcp(1.f + __expf(-x));
}
__device__ __forceinline__ float fast_tanh(float x) {
#if __has_builtin(__builtin_amdgcn_tanhf)
  return __builtin_amdgcn_tanhf(x);   // v_tanh_f32 (CDNA5 TRANS op)
#else
  return 2.f * fast_sigmoid(2.f * x) - 1.f;
#endif
}

// ---------------------------------------------------------------------------
// CDNA5 WMMA helpers (gfx1250, wave32). Layouts used here (contiguous k per lane half, same order for A and B):
//  A 16x32 f16: lane L -> row m=L%16; vgpr g, pair p -> K=(L>=16?16:0)+g*2+p
//  B 32x16 f16: lane L -> col n=L%16; K run = (L>=16?16:0) + h  (h=0..15)
//  C/D 16x16 f32: vgpr r -> M = r + (L>=16?8:0); N = L%16
// ---------------------------------------------------------------------------
__device__ __forceinline__ v8f wmma_f16(v16h a, v16h b, v8f c) {
  return __builtin_amdgcn_wmma_f32_16x16x32_f16(false, a, false, b, (short)0, c,
                                                false, false);
}

__device__ __forceinline__ v16h frag_a_lds(const _Float16* base, int stride,
                                           int kofs, int lane) {
  const int m = lane & 15;
  const int hi8 = (lane >> 4) << 4;
  const _Float16* row = base + m * stride + kofs;
  v16h a;
#pragma unroll
  for (int g = 0; g < 8; ++g) {
    const int k = hi8 + (g << 1);
    a[2 * g]     = row[k];
    a[2 * g + 1] = row[k + 1];
  }
  return a;
}

__device__ __forceinline__ v16h frag_b_lds(const _Float16* base, int stride,
                                           int lane) {
  const int n  = lane & 15;
  const int kb = (lane >> 4) << 4;
  const _Float16* row = base + n * stride + kb;
  v16h b;
#pragma unroll
  for (int h = 0; h < 16; ++h) b[h] = row[h];
  return b;
}

__device__ __forceinline__ v16h frag_b_g(const _Float16* __restrict__ W, int ldw,
                                         int nbase, int kbase, int lane) {
  const int n  = nbase + (lane & 15);
  const int kb = kbase + ((lane >> 4) << 4);
  const _Float16* p = W + (long)n * ldw + kb;
  v16h b;
#pragma unroll
  for (int h = 0; h < 16; ++h) b[h] = p[h];
  return b;
}

// ---------------------------------------------------------------------------
// Generic WMMA GEMM, fully specialized at compile time (no runtime branches):
//   AMODE: 0 = A[m*lda+k]; 1 = c0*A + c1*A2 (blend); 2 = z-permute (z_cls)
//   BMODE: 0 = weight^T (B[k][n] = W[n*ldb+k]); 1 = row-major (B[k][n] = X[k*ldb+n])
//   ACT:   relu epilogue; OUTH: write f16 output (for WMMA-consumed tensors)
// Block: 128 threads = 4 waves; tile 64(M) x 16(N); K chunked by 32.
// ---------------------------------------------------------------------------
template <int AMODE, int BMODE, bool ACT, bool OUTH>
__global__ __launch_bounds__(128) void k_gemm(
    const float* __restrict__ A, const float* __restrict__ A2, float c0, float c1,
    long lda, long aBatch,
    const float* __restrict__ Bsrc, long ldb, long bBatch,
    const float* __restrict__ bias,
    void* __restrict__ outv, long ldo, long oBatch, int K) {
  __shared__ _Float16 aT[64 * 34];
  __shared__ _Float16 bT[16 * 34];
  const int lane = threadIdx.x & 31, wave = threadIdx.x >> 5;
  const int m0 = blockIdx.x * 64, n0 = blockIdx.y * 16, bz = blockIdx.z;
  const float* Ab = A + (long)bz * aBatch;
  const float* Bb = Bsrc + (long)bz * bBatch;
  v8f acc = {};
  for (int kk = 0; kk < K; kk += 32) {
    // ---- stage A (f32 -> f16): each thread 16 consecutive floats (4x b128)
    if (AMODE == 2) {
      for (int e = threadIdx.x; e < 64 * 32; e += 128) {
        const int r = e >> 5, c = e & 31;
        const long k = kk + c, bb = k & 127, l = k >> 7;
        aT[r * 34 + c] = (_Float16)Ab[((bb << 9) + (m0 + r)) * 64 + l];
      }
    } else {
      const int r = threadIdx.x >> 1;
      const int cc = (threadIdx.x & 1) << 4;
      const float4* srcA = reinterpret_cast<const float4*>(Ab + (long)(m0 + r) * lda + kk + cc);
      _Float16* dst = &aT[r * 34 + cc];
#pragma unroll
      for (int i = 0; i < 4; ++i) {
        float4 v = srcA[i];
        if (AMODE == 1) {
          const float4 u = reinterpret_cast<const float4*>(A2 + (long)(m0 + r) * lda + kk + cc)[i];
          v.x = c0 * v.x + c1 * u.x; v.y = c0 * v.y + c1 * u.y;
          v.z = c0 * v.z + c1 * u.z; v.w = c0 * v.w + c1 * u.w;
        }
        dst[4 * i + 0] = (_Float16)v.x; dst[4 * i + 1] = (_Float16)v.y;
        dst[4 * i + 2] = (_Float16)v.z; dst[4 * i + 3] = (_Float16)v.w;
      }
    }
    // ---- stage B (f32 -> f16) as [n][k]
    if (BMODE == 0) {
      const int n = threadIdx.x >> 3;
      const int c = (threadIdx.x & 7) << 2;
      const float4 v = *reinterpret_cast<const float4*>(Bb + (long)(n0 + n) * ldb + kk + c);
      _Float16* dst = &bT[n * 34 + c];
      dst[0] = (_Float16)v.x; dst[1] = (_Float16)v.y;
      dst[2] = (_Float16)v.z; dst[3] = (_Float16)v.w;
    } else {
      const int k = threadIdx.x >> 2;
      const int n4 = (threadIdx.x & 3) << 2;
      const float4 v = *reinterpret_cast<const float4*>(Bb + (long)(kk + k) * ldb + n0 + n4);
      bT[(n4 + 0) * 34 + k] = (_Float16)v.x;
      bT[(n4 + 1) * 34 + k] = (_Float16)v.y;
      bT[(n4 + 2) * 34 + k] = (_Float16)v.z;
      bT[(n4 + 3) * 34 + k] = (_Float16)v.w;
    }
    __syncthreads();
    if (AMODE == 0 && kk + 32 < K)  // global_prefetch_b8 hint for next A chunk
      __builtin_prefetch((const void*)(Ab + (long)(m0 + (threadIdx.x >> 1)) * lda + kk + 32), 0, 3);
    const v16h af = frag_a_lds(&aT[wave * 16 * 34], 34, 0, lane);
    const v16h bf = frag_b_lds(bT, 34, lane);
    acc = wmma_f16(af, bf, acc);
    __syncthreads();
  }
  const int n = n0 + (lane & 15);
  const float bv = bias ? bias[n] : 0.f;
#pragma unroll
  for (int r = 0; r < 8; ++r) {
    const int m = m0 + wave * 16 + r + ((lane >> 4) << 3);
    float v = acc[r] + bv;
    if (ACT) v = fmaxf(v, 0.f);
    const long idx = (long)bz * oBatch + (long)m * ldo + n;
    if (OUTH) ((_Float16*)outv)[idx] = (_Float16)v;
    else      ((float*)outv)[idx] = v;
  }
}

// ---------------------------------------------------------------------------
// Encoder GRU scan (H=32). One wave per batch-tile of 16 rows; 512 steps.
// ---------------------------------------------------------------------------
__global__ __launch_bounds__(32) void k_enc_scan(
    const float* __restrict__ gi,      // [128,512,96] (includes bih)
    const _Float16* __restrict__ whh,  // [96,32] f16
    const float* __restrict__ bhh,     // [96]
    float* __restrict__ z1) {          // [128,512,32] = relu(h)
  __shared__ _Float16 hh[16 * 32];
  const int lane = threadIdx.x & 31;
  const int b0 = blockIdx.x * 16;
  const int m_lo = lane & 15, hiQ = lane >> 4;

  v16h bw[6];
#pragma unroll
  for (int g = 0; g < 6; ++g) bw[g] = frag_b_g(whh, 32, g * 16, 0, lane);

  float bR[2], bZ[2], bN[2];
#pragma unroll
  for (int tp = 0; tp < 2; ++tp) {
    const int j = tp * 16 + m_lo;
    bR[tp] = bhh[j]; bZ[tp] = bhh[32 + j]; bN[tp] = bhh[64 + j];
  }
  float hold[2][8];
#pragma unroll
  for (int tp = 0; tp < 2; ++tp)
#pragma unroll
    for (int r = 0; r < 8; ++r) hold[tp][r] = 0.f;
  for (int e = lane; e < 16 * 32; e += 32) hh[e] = (_Float16)0.f;
  __syncthreads();

  for (int t = 0; t < 512; ++t) {
    const v16h af = frag_a_lds(hh, 32, 0, lane);
    v8f accs[6];
#pragma unroll
    for (int g = 0; g < 6; ++g) {
      v8f zz = {};
      accs[g] = wmma_f16(af, bw[g], zz);
    }
#pragma unroll
    for (int tp = 0; tp < 2; ++tp) {
      const int j = tp * 16 + m_lo;
#pragma unroll
      for (int r = 0; r < 8; ++r) {
        const int m = r + (hiQ << 3);
        const long gbase = (((long)(b0 + m) << 9) + t) * 96;
        const float giR = gi[gbase + j];
        const float giZ = gi[gbase + 32 + j];
        const float giN = gi[gbase + 64 + j];
        const float rg = fast_sigmoid(giR + accs[0 + tp][r] + bR[tp]);
        const float zg = fast_sigmoid(giZ + accs[2 + tp][r] + bZ[tp]);
        const float hn = accs[4 + tp][r] + bN[tp];
        const float nn = fast_tanh(giN + rg * hn);
        const float h_new = (1.f - zg) * nn + zg * hold[tp][r];
        hold[tp][r] = h_new;
        hh[m * 32 + j] = (_Float16)h_new;  // DS in-order within wave
        z1[((((long)(b0 + m)) << 9) + t) * 32 + j] = fmaxf(h_new, 0.f);
      }
    }
    __syncthreads();
  }
}

// ---------------------------------------------------------------------------
// Decoder GRU scan (H=256, 3H=768). 8 waves share one batch-tile of 16 rows;
// wave w owns hidden cols [32w,32w+32) so gate math is lane-local.
// Per-step dec1 tile (f16, 1KB) arrives via GLOBAL_LOAD_ASYNC_TO_LDS_B128
// issued by waves 0-1, overlapped with the hh fragment gathers, completed
// with s_wait_asynccnt + workgroup barrier. 54 WMMAs / wave / step.
// ---------------------------------------------------------------------------
__global__ __launch_bounds__(256) void k_dec_scan(
    const _Float16* __restrict__ dec1, // [128,512,32] f16
    const _Float16* __restrict__ wih,  // [768,32] f16
    const _Float16* __restrict__ whh,  // [768,256] f16 (L2-resident)
    const float* __restrict__ bih, const float* __restrict__ bhh,
    float* __restrict__ xbar) {        // [128,512,256] = relu(h)
  __shared__ _Float16 hh[16 * 256];
  __shared__ _Float16 d1[16 * 32];
  const int lane = threadIdx.x & 31, wave = threadIdx.x >> 5;
  const int b0 = blockIdx.x * 16;
  const int m_lo = lane & 15, hiQ = lane >> 4;
  const int cb0 = wave * 32;

  v16h wR[2], wZ[2], wN[2];
#pragma unroll
  for (int tp = 0; tp < 2; ++tp) {
    const int cb = cb0 + tp * 16;
    wR[tp] = frag_b_g(wih, 32, cb, 0, lane);
    wZ[tp] = frag_b_g(wih, 32, 256 + cb, 0, lane);
    wN[tp] = frag_b_g(wih, 32, 512 + cb, 0, lane);
  }
  float biR[2], biZ[2], biN[2], bhR[2], bhZ[2], bhN[2];
#pragma unroll
  for (int tp = 0; tp < 2; ++tp) {
    const int j = cb0 + tp * 16 + m_lo;
    biR[tp] = bih[j];       bhR[tp] = bhh[j];
    biZ[tp] = bih[256 + j]; bhZ[tp] = bhh[256 + j];
    biN[tp] = bih[512 + j]; bhN[tp] = bhh[512 + j];
  }
  float hold[2][8];
#pragma unroll
  for (int tp = 0; tp < 2; ++tp)
#pragma unroll
    for (int r = 0; r < 8; ++r) hold[tp][r] = 0.f;
  for (int e = threadIdx.x; e < 16 * 256; e += 256) hh[e] = (_Float16)0.f;
  __syncthreads();

  // async-copy lane geometry: seg = wave*32+lane covers (row m, 8-half slice)
  const int seg  = wave * 32 + lane;        // 0..63 for waves 0-1
  const int am   = seg >> 2;                // row 0..15
  const int ak16 = (seg & 3) << 4;          // byte offset of 8-half slice in row
  const unsigned ldsD1 = (unsigned)(unsigned long)(&d1[0]) + (unsigned)(am * 64 + ak16);
  const unsigned voff  = (unsigned)(am * 32768 + ak16);  // m*16384 halfs *2B

  for (int t = 0; t < 512; ++t) {
    // issue async dec1-tile copy (1KB total): global f16 -> LDS
    if (wave < 2) {
      const _Float16* sbase = dec1 + (((long)b0) << 14) + (long)t * 32;
      asm volatile("global_load_async_to_lds_b128 %0, %1, %2"
                   :: "v"(ldsD1), "v"(voff), "s"(sbase) : "memory");
    }
    // overlap: gather hidden-state A fragments
    v16h ah[8];
#pragma unroll
    for (int kc = 0; kc < 8; ++kc) ah[kc] = frag_a_lds(hh, 256, kc * 32, lane);
    if (wave < 2) asm volatile("s_wait_asynccnt 0x0" ::: "memory");
    __syncthreads();  // d1 tile ready; everyone done reading hh

    const v16h ad = frag_a_lds(d1, 32, 0, lane);
    float hnew[2][8];
#pragma unroll
    for (int tp = 0; tp < 2; ++tp) {
      const int cb = cb0 + tp * 16;
      v8f aR = {}, aZ = {}, aNh = {}, aNi = {};
      aR  = wmma_f16(ad, wR[tp], aR);   // input-gate recompute, K=32
      aZ  = wmma_f16(ad, wZ[tp], aZ);
      aNi = wmma_f16(ad, wN[tp], aNi);
#pragma unroll
      for (int kc = 0; kc < 8; ++kc) {  // hidden matmul, K=256
        aR  = wmma_f16(ah[kc], frag_b_g(whh, 256, cb,       kc * 32, lane), aR);
        aZ  = wmma_f16(ah[kc], frag_b_g(whh, 256, 256 + cb, kc * 32, lane), aZ);
        aNh = wmma_f16(ah[kc], frag_b_g(whh, 256, 512 + cb, kc * 32, lane), aNh);
      }
      const int j = cb + m_lo;
#pragma unroll
      for (int r = 0; r < 8; ++r) {
        const int m = r + (hiQ << 3);
        const float rg = fast_sigmoid(aR[r] + biR[tp] + bhR[tp]);
        const float zg = fast_sigmoid(aZ[r] + biZ[tp] + bhZ[tp]);
        const float nn = fast_tanh(aNi[r] + biN[tp] + rg * (aNh[r] + bhN[tp]));
        const float h_new = (1.f - zg) * nn + zg * hold[tp][r];
        hold[tp][r] = h_new;
        hnew[tp][r] = h_new;
        xbar[((((long)(b0 + m)) << 9) + t) * 256 + j] = fmaxf(h_new, 0.f);
      }
    }
    __syncthreads();  // all ah/ad reads done -> safe to overwrite hh
#pragma unroll
    for (int tp = 0; tp < 2; ++tp) {
      const int j = cb0 + tp * 16 + m_lo;
#pragma unroll
      for (int r = 0; r < 8; ++r) hh[(r + (hiQ << 3)) * 256 + j] = (_Float16)hnew[tp][r];
    }
    __syncthreads();
  }
}

// ---------------------------------------------------------------------------
__global__ void k_cvt_f16(const float* __restrict__ s, _Float16* __restrict__ d, int n) {
  const int i = blockIdx.x * blockDim.x + threadIdx.x;
  if (i < n) d[i] = (_Float16)s[i];
}

__global__ void k_softmax16(const float* __restrict__ in, float* __restrict__ out, int rows) {
  const int i = blockIdx.x * blockDim.x + threadIdx.x;
  if (i >= rows) return;
  const float* r = in + (long)i * 16;
  float mx = r[0];
#pragma unroll
  for (int k = 1; k < 16; ++k) mx = fmaxf(mx, r[k]);
  float s = 0.f, e[16];
#pragma unroll
  for (int k = 0; k < 16; ++k) { e[k] = __expf(r[k] - mx); s += e[k]; }
  const float inv = fast_rcp(s);
#pragma unroll
  for (int k = 0; k < 16; ++k) out[(long)i * 16 + k] = e[k] * inv;
}

// Student-t q (V=1 -> q_k = 1/(1+d^2), row-normalized). One wave per (b,n).
__global__ __launch_bounds__(256) void k_student_q(
    const float* __restrict__ z, const float* __restrict__ cluster,
    float* __restrict__ q, int rows) {
  const int lane = threadIdx.x & 31, wave = threadIdx.x >> 5;
  const long row = (long)blockIdx.x * 8 + wave;
  if (row >= rows) return;
  const int b = (int)(row >> 9);
  const float z0 = z[row * 64 + lane];
  const float z1 = z[row * 64 + 32 + lane];
  const float* cb = cluster + (long)b * 16 * 64;
  float myq = 0.f, total = 0.f;
#pragma unroll
  for (int k = 0; k < 16; ++k) {
    const float d0 = z0 - cb[k * 64 + lane];
    const float d1 = z1 - cb[k * 64 + 32 + lane];
    float s = d0 * d0 + d1 * d1;
#pragma unroll
    for (int off = 16; off > 0; off >>= 1) s += __shfl_xor(s, off, 32);
    const float qk = fast_rcp(1.f + s);
    if (lane == k) myq = qk;
    total += qk;
  }
  if (lane < 16) q[row * 16 + lane] = myq * fast_rcp(total);
}

// ---------------------------------------------------------------------------
extern "C" void kernel_launch(void* const* d_in, const int* in_sizes, int n_in,
                              void* d_out, int out_size, void* d_ws, size_t ws_size,
                              hipStream_t stream) {
  (void)in_sizes; (void)n_in; (void)out_size; (void)ws_size;
  const float* x    = (const float*)d_in[0];
  const float* adj  = (const float*)d_in[1];
  const float* W1   = (const float*)d_in[2];
  const float* W2   = (const float*)d_in[3];
  const float* W3   = (const float*)d_in[4];
  const float* clus = (const float*)d_in[5];
  const float* eWih = (const float*)d_in[6];
  const float* eWhh = (const float*)d_in[7];
  const float* ebih = (const float*)d_in[8];
  const float* ebhh = (const float*)d_in[9];
  const float* e2W  = (const float*)d_in[10];
  const float* e2b  = (const float*)d_in[11];
  const float* muW  = (const float*)d_in[12];
  const float* mub  = (const float*)d_in[13];
  const float* d1W  = (const float*)d_in[14];
  const float* d1b  = (const float*)d_in[15];
  const float* dWih = (const float*)d_in[16];
  const float* dWhh = (const float*)d_in[17];
  const float* dbih = (const float*)d_in[18];
  const float* dbhh = (const float*)d_in[19];

  float* out  = (float*)d_out;
  float* xbar = out;                        // [128,512,256]
  float* qout = out + 16777216;             // [128,512,16]
  float* pred = out + 17825792;             // [128,512,16]
  float* zout = out + 18874368;             // [128,512,64]
  float* zcls = out + 23068672;             // [512,64]

  float* w = (float*)d_ws;
  float* enc_gi = w; w += 65536L * 96;
  float* z1     = w; w += 65536L * 32;
  _Float16* dec1h = (_Float16*)w; w += 65536L * 32 / 2;  // f16 dec1
  float* xw1    = w; w += 65536L * 32;
  float* h1     = w; w += 65536L * 32;
  float* t2     = w; w += 65536L * 64;
  float* h2     = w; w += 65536L * 64;
  float* t3     = w; w += 65536L * 16;
  float* h3     = w; w += 65536L * 16;
  _Float16* ewhh16 = (_Float16*)w; w += (96 * 32) / 2;
  _Float16* dwih16 = (_Float16*)w; w += (768 * 32) / 2;
  _Float16* dwhh16 = (_Float16*)w; w += (768 * 256) / 2;

  // weight conversions (f32 -> f16)
  k_cvt_f16<<<dim3((96 * 32 + 255) / 256), 256, 0, stream>>>(eWhh, ewhh16, 96 * 32);
  k_cvt_f16<<<dim3((768 * 32 + 255) / 256), 256, 0, stream>>>(dWih, dwih16, 768 * 32);
  k_cvt_f16<<<dim3((768 * 256 + 255) / 256), 256, 0, stream>>>(dWhh, dwhh16, 768 * 256);

  // 1) enc_gi = x @ eWih^T + ebih                          [65536,96] K=256
  k_gemm<0, 0, false, false><<<dim3(1024, 6, 1), 128, 0, stream>>>(
      x, nullptr, 0.f, 0.f, 256, 0, eWih, 256, 0, ebih, enc_gi, 96, 0, 256);
  // 2) encoder GRU scan -> z1 = relu(h)
  k_enc_scan<<<dim3(8), 32, 0, stream>>>(enc_gi, ewhh16, ebhh, z1);
  // 3) z = relu(z1 @ e2W^T + e2b)                          [65536,64] K=32
  k_gemm<0, 0, true, false><<<dim3(1024, 4, 1), 128, 0, stream>>>(
      z1, nullptr, 0.f, 0.f, 32, 0, e2W, 32, 0, e2b, zout, 64, 0, 32);
  // 4) z_cls = perm(z) @ muW^T + mub                       [512,64] K=8192
  k_gemm<2, 0, false, false><<<dim3(8, 4, 1), 128, 0, stream>>>(
      zout, nullptr, 0.f, 0.f, 0, 0, muW, 8192, 0, mub, zcls, 64, 0, 8192);
  // 5) dec1 = relu(z @ d1W^T + d1b) -> f16                 [65536,32] K=64
  k_gemm<0, 0, true, true><<<dim3(1024, 2, 1), 128, 0, stream>>>(
      zout, nullptr, 0.f, 0.f, 64, 0, d1W, 64, 0, d1b, dec1h, 32, 0, 64);
  // 6) decoder GRU scan -> x_bar = relu(h)
  k_dec_scan<<<dim3(8), 256, 0, stream>>>(dec1h, dwih16, dwhh16, dbih, dbhh, xbar);
  // 7) xw1 = x @ W1                                        [65536,32] K=256
  k_gemm<0, 1, false, false><<<dim3(1024, 2, 1), 128, 0, stream>>>(
      x, nullptr, 0.f, 0.f, 256, 0, W1, 32, 0, nullptr, xw1, 32, 0, 256);
  // 8) h1 = relu(adj @ xw1) per batch                      [512,32] K=512 x128
  k_gemm<0, 1, true, false><<<dim3(8, 2, 128), 128, 0, stream>>>(
      adj, nullptr, 0.f, 0.f, 512, 0, xw1, 32, 512L * 32, nullptr, h1, 32, 512L * 32, 512);
  // 9) t2 = (0.5*h1 + 0.5*z1) @ W2                         [65536,64] K=32
  k_gemm<1, 1, false, false><<<dim3(1024, 4, 1), 128, 0, stream>>>(
      h1, z1, 0.5f, 0.5f, 32, 0, W2, 64, 0, nullptr, t2, 64, 0, 32);
  // 10) h2 = relu(adj @ t2)                                [512,64] K=512 x128
  k_gemm<0, 1, true, false><<<dim3(8, 4, 128), 128, 0, stream>>>(
      adj, nullptr, 0.f, 0.f, 512, 0, t2, 64, 512L * 64, nullptr, h2, 64, 512L * 64, 512);
  // 11) t3 = (0.5*h2 + 0.5*z) @ W3                         [65536,16] K=64
  k_gemm<1, 1, false, false><<<dim3(1024, 1, 1), 128, 0, stream>>>(
      h2, zout, 0.5f, 0.5f, 64, 0, W3, 16, 0, nullptr, t3, 16, 0, 64);
  // 12) h3 = adj @ t3 (no relu)                            [512,16] K=512 x128
  k_gemm<0, 1, false, false><<<dim3(8, 1, 128), 128, 0, stream>>>(
      adj, nullptr, 0.f, 0.f, 512, 0, t3, 16, 512L * 16, nullptr, h3, 16, 512L * 16, 512);
  // 13) predict = softmax_K(h3)
  k_softmax16<<<dim3(256), 256, 0, stream>>>(h3, pred, 65536);
  // 14) q = student-t(z, cluster)
  k_student_q<<<dim3(8192), 256, 0, stream>>>(zout, clus, qout, 65536);
}
